// RNN_74586402062435
// MI455X (gfx1250) — matrix operand order repaired and checked by reading, not compiled, not run on hardware
//
#include <hip/hip_runtime.h>
#include <hip/hip_bf16.h>
#include <math.h>

// ---------------------------------------------------------------------------
// Problem constants (from reference): S=64, B=64, E=1024, H=1024, V=10000, L=2
// ---------------------------------------------------------------------------
#define SEQ   64
#define BATCH 64
#define EDIM  1024
#define HDIM  1024
#define VOCAB 10000
#define KCAT  2048   // E + H

typedef __attribute__((ext_vector_type(8)))  __bf16 v8bf;
typedef __attribute__((ext_vector_type(16))) __bf16 v16bf;
typedef __attribute__((ext_vector_type(8)))  float  v8f;

// ---- fp32 -> bf16 round-to-nearest-even (bit-level, NaN-safe) -------------
__device__ __forceinline__ unsigned short f32_to_bf16(float f) {
    union { float f; unsigned int u; } x;
    x.f = f;
    unsigned int r = x.u + 0x7FFFu + ((x.u >> 16) & 1u);
    if ((x.u & 0x7F800000u) == 0x7F800000u) r = x.u;   // inf/NaN: truncate
    return (unsigned short)(r >> 16);
}

// ---------------------------------------------------------------------------
// WMMA fragment loaders (CDNA5 16-bit layouts, ISA 7.12.2).
// ---------------------------------------------------------------------------
__device__ __forceinline__ v16bf load_a16(const unsigned short* p, int hi) {
    const v8bf a0 = *reinterpret_cast<const v8bf*>(p + hi * 8);
    const v8bf a1 = *reinterpret_cast<const v8bf*>(p + hi * 8 + 16);
    v16bf r;
#pragma unroll
    for (int i = 0; i < 8; ++i) { r[i] = a0[i]; r[8 + i] = a1[i]; }
    return r;
}

__device__ __forceinline__ v16bf load_b16(const unsigned short* p, int hi) {
    const v8bf b0 = *reinterpret_cast<const v8bf*>(p + hi * 8);
    const v8bf b1 = *reinterpret_cast<const v8bf*>(p + hi * 8 + 16);
    v16bf r;
#pragma unroll
    for (int i = 0; i < 8; ++i) { r[i] = b0[i]; r[8 + i] = b1[i]; }
    return r;
}

__device__ __forceinline__ v8f wmma_bf16(v16bf a, v16bf b, v8f c) {
    return __builtin_amdgcn_wmma_f32_16x16x32_bf16(
        false, a, false, b, (short)0, c, false, false);
}

// ---- CDNA5 async global->LDS copy (ASYNCcnt), inline asm ------------------
__device__ __forceinline__ void async_ld_b128(unsigned lds_off,
                                              const unsigned short* g) {
    asm volatile("global_load_async_to_lds_b128 %0, %1, off"
                 :: "v"(lds_off), "v"(g) : "memory");
}
__device__ __forceinline__ void wait_async0() {
    asm volatile("s_wait_asynccnt 0x0" ::: "memory");
}

// ---------------------------------------------------------------------------
// Kernel 1: elementwise fp32 -> bf16
// ---------------------------------------------------------------------------
__global__ void cvt_f32_bf16(const float* __restrict__ s,
                             unsigned short* __restrict__ d, int n) {
    for (int i = blockIdx.x * blockDim.x + threadIdx.x; i < n;
         i += gridDim.x * blockDim.x)
        d[i] = f32_to_bf16(s[i]);
}

// ---------------------------------------------------------------------------
// Kernel 2: embedding gather: x[s][b][e] = bf16(emb[tok[s][b]][e])
// ---------------------------------------------------------------------------
__global__ void embed_bf16(const int* __restrict__ tok,
                           const float* __restrict__ emb,
                           unsigned short* __restrict__ x, int total) {
    for (int i = blockIdx.x * blockDim.x + threadIdx.x; i < total;
         i += gridDim.x * blockDim.x) {
        const int e  = i & (EDIM - 1);
        const int sb = i >> 10;
        const int t  = tok[sb];
        x[i] = f32_to_bf16(emb[(size_t)t * EDIM + e]);
    }
}

// ---------------------------------------------------------------------------
// Pipelined 32Mx16N x Khalf GEMM body used by the RNN step (register-direct;
// the per-step GEMM is tiny and latency-bound, LDS staging not worth it).
// ---------------------------------------------------------------------------
__device__ __forceinline__ void gemm_32x16_khalf(
    const unsigned short* __restrict__ Ap0,
    const unsigned short* __restrict__ Ap1,
    const unsigned short* __restrict__ Bp,
    int hi, v8f& c0, v8f& c1) {
    v16bf a0 = load_a16(Ap0, hi);
    v16bf a1 = load_a16(Ap1, hi);
    v16bf b  = load_b16(Bp,  hi);
#pragma unroll 2
    for (int kb = 0; kb < 1024; kb += 32) {
        __builtin_prefetch(Bp + kb + 512, 0, 1);            // global_prefetch_b8
        v16bf a0n = load_a16(Ap0 + kb + 32, hi);            // harmless overread on
        v16bf a1n = load_a16(Ap1 + kb + 32, hi);            // last iter (ws-padded)
        v16bf bn  = load_b16(Bp  + kb + 32, hi);
        c0 = wmma_bf16(a0, b, c0);
        c1 = wmma_bf16(a1, b, c1);
        a0 = a0n; a1 = a1n; b = bn;
    }
}

// ---------------------------------------------------------------------------
// Kernel 3: one RNN timestep (one layer).
// ---------------------------------------------------------------------------
__global__ __launch_bounds__(256)
void rnn_step(const unsigned short* __restrict__ Xt,
              const unsigned short* __restrict__ Hp,
              const unsigned short* __restrict__ W,
              const float* __restrict__ bias,
              unsigned short* __restrict__ Hout,
              float* __restrict__ hfinal) {
    const int lane = threadIdx.x & 31;
    const int hi   = (lane >> 4) & 1;
    const int wave = threadIdx.x >> 5;
    const int w    = blockIdx.x * 8 + wave;            // 0..127
    const int nt   = w & 63;
    const int mh   = w >> 6;
    const int nbase = nt * 16;
    const int r     = lane & 15;

    const unsigned short* Bw = W + (size_t)(nbase + r) * KCAT;
    const size_t arow0 = (size_t)(mh * 32 + r) * 1024;
    const size_t arow1 = arow0 + 16 * 1024;

    v8f c0 = {}, c1 = {};
    gemm_32x16_khalf(Xt + arow0, Xt + arow1, Bw,        hi, c0, c1);
    gemm_32x16_khalf(Hp + arow0, Hp + arow1, Bw + 1024, hi, c0, c1);

    const int n  = nbase + r;
    const float bn = bias[n];
    const int mo = hi * 8;
#pragma unroll
    for (int i = 0; i < 8; ++i) {
        const int m0 = mh * 32 + mo + i;
        const float v0 = tanhf(c0[i] + bn);
        const float v1 = tanhf(c1[i] + bn);
        Hout[(size_t)m0 * HDIM + n]        = f32_to_bf16(v0);
        Hout[(size_t)(m0 + 16) * HDIM + n] = f32_to_bf16(v1);
        if (hfinal) {
            hfinal[(size_t)m0 * HDIM + n]        = v0;
            hfinal[(size_t)(m0 + 16) * HDIM + n] = v1;
        }
    }
}

// ---------------------------------------------------------------------------
// Kernel 4: logits = hs1(4096 x 1024) @ Wout^T + bout, fp32 out.
// Block tile 64M x 128N (8 waves, each 64M x 16N; 4 independent WMMAs/step).
// A (64 x 64K) and B (128 x 64K) chunks staged in LDS via async global->LDS
// copies (double buffered, ASYNCcnt + barrier), fragments read from LDS.
// Row stride padded to 72 halves (36 dwords) -> bank-conflict-free ds reads.
// Grid: 64 M-groups x 79 N-groups (last N-group: 1 of 8 waves stores).
// ---------------------------------------------------------------------------
#define KC    64    // K-chunk in halves
#define LPAD  72    // padded LDS row stride in halves (144 B, 16B-aligned)

__global__ __launch_bounds__(256)
void logits_gemm(const unsigned short* __restrict__ Abf,  // 4096 x 1024 bf16
                 const unsigned short* __restrict__ Wbf,  // 10000 x 1024 bf16
                 const float* __restrict__ bias,          // 10000 f32
                 float* __restrict__ out) {               // 4096 x 10000 f32
    __shared__ unsigned short lA[2][64 * LPAD];
    __shared__ unsigned short lB[2][128 * LPAD];

    const int tid  = threadIdx.x;
    const int lane = tid & 31;
    const int hi   = (lane >> 4) & 1;
    const int wave = tid >> 5;
    const int NG   = 79;                               // ceil(625/8)
    const int mg   = blockIdx.x / NG;                  // 0..63
    const int ng   = blockIdx.x % NG;                  // 0..78
    const int r    = lane & 15;
    const int nbase  = ng * 128 + wave * 16;           // this wave's 16 cols
    const bool active = nbase < VOCAB;

    const unsigned short* Ag = Abf + (size_t)mg * 64 * HDIM;   // 64 rows
    const unsigned short* Bg = Wbf + (size_t)ng * 128 * HDIM;  // 128 rows

    // cooperative stage of one K-chunk into LDS buffer `buf`
    auto stage = [&](int buf, int kb) {
        // A: 64 rows x 8 segs(16B) = 512 segs; 2 per thread
#pragma unroll
        for (int j = 0; j < 2; ++j) {
            const int seg = tid + 256 * j;
            const int row = seg >> 3, s = seg & 7;
            async_ld_b128((unsigned)(size_t)&lA[buf][row * LPAD + s * 8],
                          Ag + (size_t)row * HDIM + kb + s * 8);
        }
        // B: 128 rows x 8 segs = 1024 segs; 4 per thread
#pragma unroll
        for (int j = 0; j < 4; ++j) {
            const int seg = tid + 256 * j;
            const int row = seg >> 3, s = seg & 7;
            async_ld_b128((unsigned)(size_t)&lB[buf][row * LPAD + s * 8],
                          Bg + (size_t)row * HDIM + kb + s * 8);
        }
    };

    v8f c0 = {}, c1 = {}, c2 = {}, c3 = {};

    stage(0, 0);
    wait_async0();
    __syncthreads();

    for (int kb = 0; kb < HDIM; kb += KC) {
        const int buf = (kb / KC) & 1;
        if (kb + KC < HDIM) stage(buf ^ 1, kb + KC);   // uniform branch

        const unsigned short* la = &lA[buf][r * LPAD];
        const unsigned short* lb = &lB[buf][(wave * 16 + r) * LPAD];
#pragma unroll
        for (int kk = 0; kk < KC; kk += 32) {
            v16bf b  = load_b16(lb + kk, hi);
            v16bf a0 = load_a16(la + kk, hi);
            v16bf a1 = load_a16(la + 16 * LPAD + kk, hi);
            v16bf a2 = load_a16(la + 32 * LPAD + kk, hi);
            v16bf a3 = load_a16(la + 48 * LPAD + kk, hi);
            c0 = wmma_bf16(a0, b, c0);
            c1 = wmma_bf16(a1, b, c1);
            c2 = wmma_bf16(a2, b, c2);
            c3 = wmma_bf16(a3, b, c3);
        }
        wait_async0();          // this wave's stage for buf^1 complete
        __syncthreads();        // all waves done staging + done reading buf
    }

    if (active) {
        const int n  = nbase + r;
        const float bn = bias[n];
        const int mo = hi * 8;
#pragma unroll
        for (int i = 0; i < 8; ++i) {
            const size_t m = (size_t)(mg * 64 + mo + i);
            out[m * VOCAB + n]        = c0[i] + bn;
            out[(m + 16) * VOCAB + n] = c1[i] + bn;
            out[(m + 32) * VOCAB + n] = c2[i] + bn;
            out[(m + 48) * VOCAB + n] = c3[i] + bn;
        }
    }
}

// ---------------------------------------------------------------------------
// Host-side launcher
// d_in order: inputs(i32), hidden(f32), emb, W0, b0, W1, b1, Wout, bout
// d_out: logits (S,B,V) f32  then  (2,B,H) f32 final hidden
// ---------------------------------------------------------------------------
extern "C" void kernel_launch(void* const* d_in, const int* in_sizes, int n_in,
                              void* d_out, int out_size, void* d_ws, size_t ws_size,
                              hipStream_t stream) {
    (void)in_sizes; (void)n_in; (void)out_size; (void)ws_size;

    const int*   inputs = (const int*)  d_in[0];
    const float* hidden = (const float*)d_in[1];
    const float* emb    = (const float*)d_in[2];
    const float* W0     = (const float*)d_in[3];
    const float* b0     = (const float*)d_in[4];
    const float* W1     = (const float*)d_in[5];
    const float* b1     = (const float*)d_in[6];
    const float* Wout   = (const float*)d_in[7];
    const float* bout   = (const float*)d_in[8];
    float*       out    = (float*)d_out;

    // workspace carve-up (bf16), ~54.3 MB + pad
    char* ws = (char*)d_ws;
    unsigned short* W0bf   = (unsigned short*)ws;   ws += (size_t)HDIM * KCAT * 2;
    unsigned short* W1bf   = (unsigned short*)ws;   ws += (size_t)HDIM * KCAT * 2;
    unsigned short* Woutbf = (unsigned short*)ws;   ws += (size_t)VOCAB * HDIM * 2;
    unsigned short* Xbf    = (unsigned short*)ws;   ws += (size_t)SEQ * BATCH * EDIM * 2;
    unsigned short* hs0    = (unsigned short*)ws;   ws += (size_t)SEQ * BATCH * HDIM * 2;
    unsigned short* hs1    = (unsigned short*)ws;   ws += (size_t)SEQ * BATCH * HDIM * 2;
    unsigned short* hinit  = (unsigned short*)ws;   ws += (size_t)2 * BATCH * HDIM * 2;
    ws += 4096;                                     // pipeline-overread pad

    // 1) weight / initial-hidden conversion to bf16
    cvt_f32_bf16<<<2048, 256, 0, stream>>>(W0,     W0bf,   HDIM * KCAT);
    cvt_f32_bf16<<<2048, 256, 0, stream>>>(W1,     W1bf,   HDIM * KCAT);
    cvt_f32_bf16<<<4096, 256, 0, stream>>>(Wout,   Woutbf, VOCAB * HDIM);
    cvt_f32_bf16<<<512,  256, 0, stream>>>(hidden, hinit,  2 * BATCH * HDIM);

    // 2) embedding gather -> bf16 activations
    embed_bf16<<<4096, 256, 0, stream>>>(inputs, emb, Xbf, SEQ * BATCH * EDIM);

    const size_t stepElems = (size_t)BATCH * HDIM;
    float* hfin0 = out + (size_t)SEQ * BATCH * VOCAB;   // h0T
    float* hfin1 = hfin0 + stepElems;                   // h1T

    // 3) layer 0: sequential over time
    for (int t = 0; t < SEQ; ++t) {
        const unsigned short* Xt = Xbf + (size_t)t * BATCH * EDIM;
        const unsigned short* Hp = (t == 0) ? hinit : hs0 + (size_t)(t - 1) * stepElems;
        rnn_step<<<16, 256, 0, stream>>>(Xt, Hp, W0bf, b0,
                                         hs0 + (size_t)t * stepElems,
                                         (t == SEQ - 1) ? hfin0 : (float*)nullptr);
    }
    // 4) layer 1: sequential over time, consumes hs0
    for (int t = 0; t < SEQ; ++t) {
        const unsigned short* Xt = hs0 + (size_t)t * stepElems;
        const unsigned short* Hp = (t == 0) ? hinit + stepElems
                                            : hs1 + (size_t)(t - 1) * stepElems;
        rnn_step<<<16, 256, 0, stream>>>(Xt, Hp, W1bf, b1,
                                         hs1 + (size_t)t * stepElems,
                                         (t == SEQ - 1) ? hfin1 : (float*)nullptr);
    }

    // 5) output projection: 4096 x 10000 logits, LDS-staged async GEMM
    logits_gemm<<<64 * 79, 256, 0, stream>>>(hs1, Woutbf, bout, out);
}
